// SADModel_81114752352526
// MI455X (gfx1250) — compile-verified, matrix operand order checked by reading, not run on hardware
//
#include <hip/hip_runtime.h>
#include <hip/hip_bf16.h>
#include <stdint.h>

// Problem constants (reference: H=128, D_IN=40, B=32, T=2000)
#define HID   128
#define GH    512            // 4*H gate width
#define BATCH 32
#define TLEN  2000
#define ROWS  (BATCH * TLEN) // 64000 rows; t-major: row = t*32 + b
#define NTILES (GH / 16)     // 32 N-tiles of the gate dimension

typedef __attribute__((ext_vector_type(16))) __bf16 bf16x16;
typedef __attribute__((ext_vector_type(8)))  float  f32x8;

union Frag {
    bf16x16        bf;
    uint4          u4[2];
    unsigned short us[16];
};
union AccU {
    f32x8  v;
    float4 f4[2];
};

__device__ __forceinline__ unsigned short f32_to_bf16_rne(float f) {
    union { float f; uint32_t u; } a; a.f = f;
    uint32_t r = a.u + 0x7FFFu + ((a.u >> 16) & 1u);
    return (unsigned short)(r >> 16);
}

// gfx1250 has a hardware V_TANH_F32 (TRANS op -> co-executes with XDL WMMA).
#if __has_builtin(__builtin_amdgcn_tanhf)
__device__ __forceinline__ float fast_tanh(float x) { return __builtin_amdgcn_tanhf(x); }
#else
__device__ __forceinline__ float fast_tanh(float x) { return tanhf(x); }
#endif
__device__ __forceinline__ float sigm(float x) { return 0.5f * fast_tanh(0.5f * x) + 0.5f; }

// CDNA5 16-bit A/B WMMA fragment K-pattern (ISA 7.12.2) as two contiguous
// 8-element runs per lane:
//   us[e]   (e<8) <- k = base + lhi*8 + e
//   us[e+8] (e<8) <- k = base + 16 + lhi*8 + e

// ---------------------------------------------------------------------------
// Pack a row-major f32 weight W[N=512][KREAL] into bf16 WMMA B-fragments:
// dst[(kt*NTILES + nt)*32 + lane][16].  K zero-padded to 32*ktiles.
// grid = (NTILES, ktiles), block = 32 (one wave per fragment).
// ---------------------------------------------------------------------------
template <int KREAL>
__global__ void pack_bfrag(const float* __restrict__ W,
                           unsigned short* __restrict__ dst) {
    const int lane = threadIdx.x;
    const int lm = lane & 15, lhi = lane >> 4;
    const int nt = blockIdx.x, kt = blockIdx.y;
    const int n = nt * 16 + lm;
    unsigned short* d = dst + ((size_t)(kt * gridDim.x + nt) * 32 + lane) * 16;
#pragma unroll
    for (int half = 0; half < 2; ++half) {
        const int kbase = kt * 32 + half * 16 + lhi * 8;
#pragma unroll
        for (int e = 0; e < 8; ++e) {
            const int k = kbase + e;
            float v = (k < KREAL) ? W[(size_t)n * KREAL + k] : 0.0f;
            d[half * 8 + e] = f32_to_bf16_rne(v);
        }
    }
}

__global__ void combine_bias(const float* __restrict__ a, const float* __restrict__ b,
                             float* __restrict__ o, int n) {
    int i = blockIdx.x * blockDim.x + threadIdx.x;
    if (i < n) o[i] = a[i] + b[i];
}

// ---------------------------------------------------------------------------
// xg = A @ W^T + bias, rows in t-major order (row = t*32 + b).  Output is
// written RAW in C/D fragment order: C[(tile16*NTILES + ntile)*256 + lane*8
// .. +8], so the scan ingests accumulators with global_load_b128 pairs.
// XMAJ=true remaps t-major rows onto the b-major x tensor.
// grid.x = rows/16; block = 256 (8 waves x 4 N-tiles each).
// ---------------------------------------------------------------------------
template <int KREAL, int KTILES, bool XMAJ>
__global__ __launch_bounds__(256)
void wmma_xgemm(const float* __restrict__ A,
                const unsigned short* __restrict__ Bfrag,
                const float* __restrict__ bias,
                float* __restrict__ C) {
    const int wave = threadIdx.x >> 5;
    const int lane = threadIdx.x & 31;
    const int lm = lane & 15, lhi = lane >> 4;
    const int row = blockIdx.x * 16 + lm;                 // t-major row for this lane
    const float* __restrict__ Arow =
        XMAJ ? A + ((size_t)(row & 31) * TLEN + (row >> 5)) * KREAL
             : A + (size_t)row * KREAL;

    f32x8 acc[4] = {};
#pragma unroll
    for (int kt = 0; kt < KTILES; ++kt) {
        Frag af;  // two contiguous 8-float runs, f32 -> bf16 (K tail folds to 0)
#pragma unroll
        for (int half = 0; half < 2; ++half) {
            const int kbase = kt * 32 + half * 16 + lhi * 8;
#pragma unroll
            for (int e = 0; e < 8; ++e) {
                float v = (kbase + e < KREAL) ? Arow[kbase + e] : 0.0f;
                af.us[half * 8 + e] = f32_to_bf16_rne(v);
            }
        }
#pragma unroll
        for (int nt = 0; nt < 4; ++nt) {
            const int ntile = wave * 4 + nt;
            Frag bf;
            const uint4* bsrc =
                (const uint4*)(Bfrag + ((size_t)(kt * NTILES + ntile) * 32 + lane) * 16);
            bf.u4[0] = bsrc[0];
            bf.u4[1] = bsrc[1];
            acc[nt] = __builtin_amdgcn_wmma_f32_16x16x32_bf16(
                false, af.bf, false, bf.bf, (short)0, acc[nt], false, false);
        }
    }
#pragma unroll
    for (int nt = 0; nt < 4; ++nt) {
        const int ntile = wave * 4 + nt;
        const float bv = bias[ntile * 16 + lm];
        float* p = C + ((size_t)blockIdx.x * NTILES + ntile) * 256 + lane * 8;
        float4 lo, hi;
        lo.x = acc[nt][0] + bv; lo.y = acc[nt][1] + bv;
        lo.z = acc[nt][2] + bv; lo.w = acc[nt][3] + bv;
        hi.x = acc[nt][4] + bv; hi.y = acc[nt][5] + bv;
        hi.z = acc[nt][6] + bv; hi.w = acc[nt][7] + bv;
        *(float4*)(p)     = lo;
        *(float4*)(p + 4) = hi;
    }
}

// ---------------------------------------------------------------------------
// Serial LSTM scan, one workgroup per direction (blockIdx.x: 0=fwd, 1=bwd).
//  - W_hh B-fragments pinned in VGPRs (16 frags = 128 regs/lane, loop-invariant)
//  - Wave w owns gate tiles {w, w+8, w+16, w+24} == same 16 hidden columns of
//    i,f,g,o -> the cell update is fused in registers, no gate exchange in LDS
//  - h carried in a double-buffered bf16 LDS buffer -> ONE barrier per step
//  - xg ingested as raw accumulator fragments (2 x b128 per tile)
// Dynamic LDS: 2 * 8 KB.
// ---------------------------------------------------------------------------
__global__ __launch_bounds__(256)
void lstm_scan(const float* __restrict__ xg_f, const float* __restrict__ xg_b,
               const unsigned short* __restrict__ wfrag_f,
               const unsigned short* __restrict__ wfrag_b,
               float* __restrict__ out, int T) {
    extern __shared__ char smem[];
    unsigned short* hbuf = (unsigned short*)smem;   // [2][32][128] bf16 (16 KB)

    const int dir = blockIdx.x;
    const float*          xg    = dir ? xg_b : xg_f;
    const unsigned short* wfrag = dir ? wfrag_b : wfrag_f;

    const int tid = threadIdx.x;
    const int wave = tid >> 5, lane = tid & 31;
    const int lm = lane & 15, lhi = lane >> 4;
    const int jcol = wave * 16 + lm;   // hidden column owned by this lane

    // Pin all W_hh B-fragments this wave needs in VGPRs: breg[kt][gate],
    // gate g uses ntile = wave + 8*g (columns jcol of gate g).
    Frag breg[4][4];
#pragma unroll
    for (int kt = 0; kt < 4; ++kt)
#pragma unroll
        for (int g = 0; g < 4; ++g) {
            const uint4* p = (const uint4*)(wfrag +
                ((size_t)(kt * NTILES + wave + 8 * g) * 32 + lane) * 16);
            breg[kt][g].u4[0] = p[0];
            breg[kt][g].u4[1] = p[1];
        }

    for (int i = tid; i < 2 * BATCH * HID / 8; i += 256)
        ((uint4*)hbuf)[i] = make_uint4(0, 0, 0, 0);
    __syncthreads();

    // Cell state: c[(mt*8 + r)] for (b = mt*16 + r + lhi*8, j = jcol)
    float c[16];
#pragma unroll
    for (int q = 0; q < 16; ++q) c[q] = 0.0f;

    for (int s = 0; s < T; ++s) {
        const int t = dir ? (T - 1 - s) : s;
        const unsigned short* hbA = hbuf + (s & 1) * (BATCH * HID);       // read h(t-1)
        unsigned short*       hbN = hbuf + ((s + 1) & 1) * (BATCH * HID); // write h(t)

        if (s + 1 < T) {  // prefetch next step's xg fragments
            const int tn = dir ? (T - 2 - s) : (s + 1);
            __builtin_prefetch(xg + ((size_t)(tn * 2) * NTILES + wave) * 256 + lane * 8, 0, 1);
            __builtin_prefetch(xg + ((size_t)(tn * 2 + 1) * NTILES + wave) * 256 + lane * 8, 0, 1);
        }

#pragma unroll
        for (int mt = 0; mt < 2; ++mt) {
            // accumulators = xg fragments (raw layout) for the 4 gates
            AccU acc[4];
#pragma unroll
            for (int g = 0; g < 4; ++g) {
                const float* p =
                    xg + ((size_t)(t * 2 + mt) * NTILES + wave + 8 * g) * 256 + lane * 8;
                acc[g].f4[0] = *(const float4*)(p);
                acc[g].f4[1] = *(const float4*)(p + 4);
            }
            // g += h(t-1) @ W_hh^T : 16 WMMAs, B operands resident in VGPRs
#pragma unroll
            for (int kt = 0; kt < 4; ++kt) {
                Frag af;  // A fragment = two 16-byte LDS reads of h(t-1)
                const unsigned short* hrow =
                    hbA + (mt * 16 + lm) * HID + kt * 32 + lhi * 8;
                af.u4[0] = *(const uint4*)(hrow);
                af.u4[1] = *(const uint4*)(hrow + 16);
#pragma unroll
                for (int g = 0; g < 4; ++g)
                    acc[g].v = __builtin_amdgcn_wmma_f32_16x16x32_bf16(
                        false, af.bf, false, breg[kt][g].bf, (short)0, acc[g].v,
                        false, false);
            }
            // fused gates + cell update, entirely in registers
#pragma unroll
            for (int r = 0; r < 8; ++r) {
                const int ci = mt * 8 + r;
                const int b  = mt * 16 + r + lhi * 8;
                const float gi = sigm(acc[0].v[r]);
                const float gf = sigm(acc[1].v[r]);
                const float gg = fast_tanh(acc[2].v[r]);
                const float go = sigm(acc[3].v[r]);
                c[ci] = gf * c[ci] + gi * gg;
                const float h = go * fast_tanh(c[ci]);
                hbN[b * HID + jcol] = f32_to_bf16_rne(h);
                out[((size_t)t * BATCH + b) * (2 * HID) + dir * HID + jcol] = h;
            }
        }
        __syncthreads();  // publish h(t) before anyone reads it at step s+1
    }
}

// ---------------------------------------------------------------------------
// Final FC: out[b*T + t] = dot(h1[t-major row r=t*32+b, :256], fc_w) + fc_b.
// One wave per row.
// ---------------------------------------------------------------------------
__global__ __launch_bounds__(256)
void fc_dot(const float* __restrict__ Hin, const float* __restrict__ w,
            const float* __restrict__ bptr, float* __restrict__ out, int rows) {
    const int wave = threadIdx.x >> 5, lane = threadIdx.x & 31;
    const int row = blockIdx.x * 8 + wave;
    if (row >= rows) return;
    const float* hr = Hin + (size_t)row * (2 * HID);
    float v = 0.0f;
#pragma unroll
    for (int q = 0; q < 8; ++q) v += hr[lane + 32 * q] * w[lane + 32 * q];
#pragma unroll
    for (int off = 16; off; off >>= 1) v += __shfl_xor(v, off, 32);
    if (lane == 0) out[(size_t)(row & 31) * TLEN + (row >> 5)] = v + bptr[0];
}

// ---------------------------------------------------------------------------
extern "C" void kernel_launch(void* const* d_in, const int* in_sizes, int n_in,
                              void* d_out, int out_size, void* d_ws, size_t ws_size,
                              hipStream_t stream) {
    (void)in_sizes; (void)n_in; (void)out_size; (void)ws_size;
    const float* x      = (const float*)d_in[0];
    // order: [l0 fwd, l0 bwd, l1 fwd, l1 bwd]
    const float* wih[4] = { (const float*)d_in[1],  (const float*)d_in[5],
                            (const float*)d_in[9],  (const float*)d_in[13] };
    const float* whh[4] = { (const float*)d_in[2],  (const float*)d_in[6],
                            (const float*)d_in[10], (const float*)d_in[14] };
    const float* bih[4] = { (const float*)d_in[3],  (const float*)d_in[7],
                            (const float*)d_in[11], (const float*)d_in[15] };
    const float* bhh[4] = { (const float*)d_in[4],  (const float*)d_in[8],
                            (const float*)d_in[12], (const float*)d_in[16] };
    const float* fcw = (const float*)d_in[17];
    const float* fcb = (const float*)d_in[18];
    float* outp = (float*)d_out;

    char* ws = (char*)d_ws;
    size_t off = 0;
    auto alloc = [&](size_t sz) -> char* {
        char* p = ws + off;
        off += (sz + 255) & ~(size_t)255;
        return p;
    };

    const int kt_ih[4] = { 2, 2, 8, 8 };  // K=40 padded to 64; K=256

    unsigned short* wihfrag[4];
    unsigned short* whhfrag[4];
    float*          biasp[4];
    for (int i = 0; i < 4; ++i)
        wihfrag[i] = (unsigned short*)alloc((size_t)kt_ih[i] * NTILES * 32 * 16 * 2);
    for (int i = 0; i < 4; ++i)
        whhfrag[i] = (unsigned short*)alloc((size_t)4 * NTILES * 32 * 16 * 2);
    for (int i = 0; i < 4; ++i)
        biasp[i] = (float*)alloc(GH * sizeof(float));

    float* xgA = (float*)alloc((size_t)ROWS * GH * sizeof(float));      // frag layout
    float* xgB = (float*)alloc((size_t)ROWS * GH * sizeof(float));      // frag layout
    float* h0  = (float*)alloc((size_t)ROWS * 2 * HID * sizeof(float)); // t-major
    float* h1  = (float*)alloc((size_t)ROWS * 2 * HID * sizeof(float)); // t-major

    // ---- weight packing (bf16 WMMA fragment layout) ----
    for (int i = 0; i < 4; ++i)
        combine_bias<<<2, 256, 0, stream>>>(bih[i], bhh[i], biasp[i], GH);
    pack_bfrag<40><<<dim3(NTILES, 2), 32, 0, stream>>>(wih[0], wihfrag[0]);
    pack_bfrag<40><<<dim3(NTILES, 2), 32, 0, stream>>>(wih[1], wihfrag[1]);
    pack_bfrag<256><<<dim3(NTILES, 8), 32, 0, stream>>>(wih[2], wihfrag[2]);
    pack_bfrag<256><<<dim3(NTILES, 8), 32, 0, stream>>>(wih[3], wihfrag[3]);
    for (int i = 0; i < 4; ++i)
        pack_bfrag<HID><<<dim3(NTILES, 4), 32, 0, stream>>>(whh[i], whhfrag[i]);

    const size_t scan_lds = 2 * BATCH * HID * 2;  // 16 KB double-buffered h

    // ---- layer 0: input GEMMs (parallel), then serial scan ----
    wmma_xgemm<40, 2, true><<<ROWS / 16, 256, 0, stream>>>(x, wihfrag[0], biasp[0], xgA);
    wmma_xgemm<40, 2, true><<<ROWS / 16, 256, 0, stream>>>(x, wihfrag[1], biasp[1], xgB);
    lstm_scan<<<2, 256, scan_lds, stream>>>(xgA, xgB, whhfrag[0], whhfrag[1], h0, TLEN);

    // ---- layer 1 (input dim 256, h0 already t-major), reusing xg scratch ----
    wmma_xgemm<256, 8, false><<<ROWS / 16, 256, 0, stream>>>(h0, wihfrag[2], biasp[2], xgA);
    wmma_xgemm<256, 8, false><<<ROWS / 16, 256, 0, stream>>>(h0, wihfrag[3], biasp[3], xgB);
    lstm_scan<<<2, 256, scan_lds, stream>>>(xgA, xgB, whhfrag[2], whhfrag[3], h1, TLEN);

    // ---- FC head ----
    fc_dot<<<(ROWS + 7) / 8, 256, 0, stream>>>(h1, fcw, fcb, outp, ROWS);
}
